// GNNLinkPredictor_10694468567086
// MI455X (gfx1250) — matrix operand order repaired and checked by reading, not compiled, not run on hardware
//
#include <hip/hip_runtime.h>

#define NN 100000
#define NE 640000
#define NL 100000
#define HIDC 128
#define OUTC 64

typedef __attribute__((ext_vector_type(16))) __bf16          v16bf;
typedef __attribute__((ext_vector_type(8)))  float           v8f;
typedef __attribute__((ext_vector_type(16))) unsigned short  v16us;
typedef __attribute__((ext_vector_type(8)))  unsigned short  v8us;

// round-to-nearest-even f32 -> bf16 bits (used only in memory-bound passes)
__device__ __forceinline__ unsigned short f2bf(float f) {
    unsigned u = __builtin_bit_cast(unsigned, f);
    u += 0x7FFFu + ((u >> 16) & 1u);
    return (unsigned short)(u >> 16);
}
__device__ __forceinline__ float bf2f(unsigned short b) {
    unsigned u = ((unsigned)b) << 16;
    return __builtin_bit_cast(float, u);
}

// ---- fragment loaders: pure b128 vector loads, no conversion ----
// A fragment: lane holds row m, K = kb+{0..7} and kb+{16..23} (kb = k0 + 8*half)
__device__ __forceinline__ v16bf ld_frag_a(const unsigned short* p) {
    v8us lo = *(const v8us*)(p);
    v8us hi = *(const v8us*)(p + 16);
    v16us f = __builtin_shufflevector(lo, hi, 0,1,2,3,4,5,6,7,8,9,10,11,12,13,14,15);
    return __builtin_bit_cast(v16bf, f);
}
// B fragment: lane holds col n, K = kb+{0..7} and kb+{16..23} (kb = k0 + 8*half)
__device__ __forceinline__ v16bf ld_frag_b(const unsigned short* p) {
    v8us lo = *(const v8us*)(p);
    v8us hi = *(const v8us*)(p + 16);
    v16us f = __builtin_shufflevector(lo, hi, 0,1,2,3,4,5,6,7,8,9,10,11,12,13,14,15);
    return __builtin_bit_cast(v16bf, f);
}

__global__ void zero_f32(float* __restrict__ p, int n) {
    int i = blockIdx.x * blockDim.x + threadIdx.x;
    if (i < n) p[i] = 0.0f;
}

// f32 -> bf16 elementwise (weights, x)
__global__ void conv_bf16(const float* __restrict__ in, unsigned short* __restrict__ out, int n) {
    int i = blockIdx.x * blockDim.x + threadIdx.x;
    if (i < n) out[i] = f2bf(in[i]);
}

// meanbf[i] = bf16(agg[i] / max(cnt[row],1))
__global__ void mean_bf16(const float* __restrict__ agg, const float* __restrict__ cnt,
                          unsigned short* __restrict__ meanbf, long long n) {
    long long i = (long long)blockIdx.x * blockDim.x + threadIdx.x;
    if (i >= n) return;
    float rc = 1.0f / fmaxf(cnt[(int)(i >> 7)], 1.0f);
    meanbf[i] = f2bf(agg[i] * rc);
}

// one thread per (edge, channel): agg[dst][c] += feat[src][c]
template <bool SRC_BF16>
__global__ void scatter_feat(const void* __restrict__ feat,
                             const int* __restrict__ src,
                             const int* __restrict__ dst,
                             float* __restrict__ agg, int nE) {
    long long i = (long long)blockIdx.x * blockDim.x + threadIdx.x;
    long long total = (long long)nE * 128;
    if (i >= total) return;
    int e = (int)(i >> 7);
    int c = (int)(i & 127);
    int s = src[e], d = dst[e];
    float v;
    if (SRC_BF16) v = bf2f(((const unsigned short*)feat)[(long long)s * 128 + c]);
    else          v = ((const float*)feat)[(long long)s * 128 + c];
    __hip_atomic_fetch_add(&agg[(long long)d * 128 + c], v,
                           __ATOMIC_RELAXED, __HIP_MEMORY_SCOPE_AGENT);
}

__global__ void count_edges(const int* __restrict__ dst,
                            float* __restrict__ cnt, int nE) {
    int e = blockIdx.x * blockDim.x + threadIdx.x;
    if (e < nE)
        __hip_atomic_fetch_add(&cnt[dst[e]], 1.0f,
                               __ATOMIC_RELAXED, __HIP_MEMORY_SCOPE_AGENT);
}

// out[m][n] = act( mean[m,:]@Wl[n,:] + xin[m,:]@Wr[n,:] + bias[n] )
// One wave per 16x16 output tile; B-fragments kept in registers across MT m-tiles.
#define MT 5
template <bool RELU, bool OUT_BF16>
__global__ __launch_bounds__(32)
void sage_wmma(const unsigned short* __restrict__ meanbf, // [NN,128] bf16
               const unsigned short* __restrict__ xbf,    // [NN,128] bf16
               const unsigned short* __restrict__ Wlbf,   // [nOut,128] bf16
               const unsigned short* __restrict__ Wrbf,   // [nOut,128] bf16
               const float* __restrict__ bias,            // [nOut]
               void* __restrict__ outv,                   // [NN,nOut]
               int nOut) {
    const int lane = threadIdx.x;
    const int half = lane >> 4;
    const int l16  = lane & 15;
    const int ncol = blockIdx.y * 16 + l16;

    // Preload all B fragments (4 K-steps x 2 weight matrices) into registers.
    v16bf Bl[4], Br[4];
    const unsigned short* wl = Wlbf + (long long)ncol * 128;
    const unsigned short* wr = Wrbf + (long long)ncol * 128;
    #pragma unroll
    for (int kk = 0; kk < 4; ++kk) {
        const int kbB = kk * 32 + 8 * half;
        Bl[kk] = ld_frag_b(wl + kbB);
        Br[kk] = ld_frag_b(wr + kbB);
    }
    const float bv = bias[ncol];

    #pragma unroll
    for (int t = 0; t < MT; ++t) {
        const int m0  = (blockIdx.x * MT + t) * 16;
        const int row = m0 + l16;
        const unsigned short* mrow = meanbf + (long long)row * 128;
        const unsigned short* xrow = xbf    + (long long)row * 128;

        v8f c = {};
        #pragma unroll
        for (int kk = 0; kk < 4; ++kk) {
            const int kbA = kk * 32 + 8 * half;
            v16bf am = ld_frag_a(mrow + kbA);
            v16bf ax = ld_frag_a(xrow + kbA);
            c = __builtin_amdgcn_wmma_f32_16x16x32_bf16(false, am, false, Bl[kk],
                                                        (short)0, c, false, false);
            c = __builtin_amdgcn_wmma_f32_16x16x32_bf16(false, ax, false, Br[kk],
                                                        (short)0, c, false, false);
        }
        // C/D layout: VGPR r holds M = m0 + r + 8*half, N = ncol
        #pragma unroll
        for (int r = 0; r < 8; ++r) {
            const int m = m0 + r + 8 * half;
            float v = c[r] + bv;
            if (RELU) v = fmaxf(v, 0.0f);
            if (OUT_BF16) ((unsigned short*)outv)[(long long)m * nOut + ncol] = f2bf(v);
            else          ((float*)outv)[(long long)m * nOut + ncol] = v;
        }
    }
}

// scores[i] = dot64(z[a[i]], z[b[i]]) : one wave per label edge
__global__ void decode_dot(const float* __restrict__ z,
                           const int* __restrict__ a,
                           const int* __restrict__ b,
                           float* __restrict__ scores, int nL) {
    int w    = (int)((blockIdx.x * (long long)blockDim.x + threadIdx.x) >> 5);
    int lane = threadIdx.x & 31;
    if (w >= nL) return;
    const float* za = z + (long long)a[w] * 64;
    const float* zb = z + (long long)b[w] * 64;
    float p = za[lane] * zb[lane] + za[lane + 32] * zb[lane + 32];
    #pragma unroll
    for (int off = 16; off > 0; off >>= 1) p += __shfl_xor(p, off, 32);
    if (lane == 0) scores[w] = p;
}

extern "C" void kernel_launch(void* const* d_in, const int* in_sizes, int n_in,
                              void* d_out, int out_size, void* d_ws, size_t ws_size,
                              hipStream_t stream) {
    const float* x   = (const float*)d_in[0];
    const int*   ei  = (const int*)d_in[1];
    const int*   eli = (const int*)d_in[2];
    const float* W1l = (const float*)d_in[3];
    const float* b1  = (const float*)d_in[4];
    const float* W1r = (const float*)d_in[5];
    const float* W2l = (const float*)d_in[6];
    const float* b2  = (const float*)d_in[7];
    const float* W2r = (const float*)d_in[8];
    float* scores = (float*)d_out;

    char* ws = (char*)d_ws;
    float*          agg    = (float*)(ws);                            // NN*128 f32
    float*          cnt    = (float*)(ws + (size_t)51200000);         // NN f32
    unsigned short* xbf    = (unsigned short*)(ws + (size_t)51600000);// NN*128 bf16
    unsigned short* hbf    = (unsigned short*)(ws + (size_t)77200000);// NN*128 bf16
    unsigned short* meanbf = (unsigned short*)(ws + (size_t)102800000);// NN*128 bf16
    float*          z      = (float*)(ws + (size_t)128400000);        // NN*64 f32
    unsigned short* w1lb   = (unsigned short*)(ws + (size_t)154000000);// 16384
    unsigned short* w1rb   = w1lb + 16384;
    unsigned short* w2lb   = w1rb + 16384;
    unsigned short* w2rb   = w2lb + 8192;

    const int* src = ei;
    const int* dst = ei + NE;
    const int* la  = eli;
    const int* lb  = eli + NL;

    const long long nFeat = (long long)NN * 128;
    const int featBlocks  = (int)((nFeat + 255) / 256);
    const long long nScat = (long long)NE * 128;
    const int scatBlocks  = (int)((nScat + 255) / 256);

    // weight + x conversion (bf16)
    conv_bf16<<<(16384 + 255) / 256, 256, 0, stream>>>(W1l, w1lb, 16384);
    conv_bf16<<<(16384 + 255) / 256, 256, 0, stream>>>(W1r, w1rb, 16384);
    conv_bf16<<<(8192  + 255) / 256, 256, 0, stream>>>(W2l, w2lb, 8192);
    conv_bf16<<<(8192  + 255) / 256, 256, 0, stream>>>(W2r, w2rb, 8192);
    conv_bf16<<<featBlocks, 256, 0, stream>>>(x, xbf, (int)nFeat);

    // layer 1 aggregation (agg + cnt contiguous)
    zero_f32<<<(NN * 129 + 255) / 256, 256, 0, stream>>>(agg, NN * 129);
    scatter_feat<false><<<scatBlocks, 256, 0, stream>>>(x, src, dst, agg, NE);
    count_edges<<<(NE + 255) / 256, 256, 0, stream>>>(dst, cnt, NE);
    mean_bf16<<<featBlocks, 256, 0, stream>>>(agg, cnt, meanbf, nFeat);

    // layer 1 GEMM + relu -> hbf (bf16)
    {
        dim3 g(NN / 16 / MT, HIDC / 16);
        sage_wmma<true, true><<<g, 32, 0, stream>>>(meanbf, xbf, w1lb, w1rb, b1, hbf, HIDC);
    }

    // layer 2 aggregation (reuse cnt)
    zero_f32<<<(NN * 128 + 255) / 256, 256, 0, stream>>>(agg, NN * 128);
    scatter_feat<true><<<scatBlocks, 256, 0, stream>>>(hbf, src, dst, agg, NE);
    mean_bf16<<<featBlocks, 256, 0, stream>>>(agg, cnt, meanbf, nFeat);

    // layer 2 GEMM -> z (f32)
    {
        dim3 g(NN / 16 / MT, OUTC / 16);
        sage_wmma<false, false><<<g, 32, 0, stream>>>(meanbf, hbf, w2lb, w2rb, b2, z, OUTC);
    }

    // edge dot-product decode
    decode_dot<<<((long long)NL * 32 + 255) / 256, 256, 0, stream>>>(z, la, lb, scores, NL);
}
